// MesoNet_62646392979830
// MI455X (gfx1250) — matrix operand order repaired and checked by reading, not compiled, not run on hardware
//
#include <hip/hip_runtime.h>
#include <hip/hip_bf16.h>
#include <math.h>

// MesoNet forward for gfx1250 (MI455X). All dense GEMMs go through a
// wave32 WMMA kernel (v_wmma_f32_16x16x32_f16, f32 accumulate). Graph
// segment reductions are deterministic (one workgroup per graph, serial
// over that graph's edges/nodes in LDS) -- no float atomics anywhere.
// NNConv edge staging uses GLOBAL_LOAD_ASYNC_TO_LDS_B32 (ASYNCcnt path).

typedef __attribute__((ext_vector_type(16))) _Float16 v16h;
typedef __attribute__((ext_vector_type(8)))  float    v8f;

#define ACT_NONE  0
#define ACT_RELU  1
#define ACT_TANH  2
#define ACT_SIG   3
#define ACT_LECUN 4

__device__ __forceinline__ float apply_act(float v, int act) {
  switch (act) {
    case ACT_RELU:  return fmaxf(v, 0.0f);
    case ACT_TANH:  return tanhf(v);
    case ACT_SIG:   return 1.0f / (1.0f + expf(-v));
    case ACT_LECUN: return 1.7159f * tanhf(0.666667f * v);
    default:        return v;
  }
}
__device__ __forceinline__ int imin(int a, int b) { return a < b ? a : b; }

// ---------------------------------------------------------------------------
// WMMA GEMM: C[M,N] = act( A[M,K](lda) @ W[K,N] + bias (+ C if accum) )
// One wave per 16x16 tile, K-loop in steps of 32.
// Tiles are staged in LDS *directly in WMMA fragment order* so each lane
// fetches its v16h fragment as one aligned 32B LDS read (2x ds_load_b128).
// Global loads are branch-free: clamped address + select-zero (no exec-mask
// branches, no per-element waits).
// Fragment layouts per CDNA5 ISA 7.12.2 (wave32):
//   A (16x32 f16): lane = m + 16*k[3], element e = (k>>4)*8 + (k&7)
//   B (32x16 f16): lane = n + 16*k[3], element e = (k>>4)*8 + (k&7)
//   C/D (16x16 f32): reg j -> m = j + 8*(lane>>4), n = lane&15
// ---------------------------------------------------------------------------
__global__ __launch_bounds__(32) void gemm_wmma_kernel(
    const float* __restrict__ A, int lda,
    const float* __restrict__ W,
    const float* __restrict__ bias,
    float* __restrict__ C,
    int M, int N, int K, int act, int accum)
{
  __shared__ _Float16 Af[32 * 16] __attribute__((aligned(32)));
  __shared__ _Float16 Bf[32 * 16] __attribute__((aligned(32)));
  const int lane = threadIdx.x;
  const int row0 = blockIdx.y * 16;
  const int col0 = blockIdx.x * 16;
  v8f acc = {};
  for (int k0 = 0; k0 < K; k0 += 32) {
    // ---- stage A tile: 16 coalesced rounds, one row of 32 floats each ----
    {
      const int gk   = k0 + lane;              // this lane's k within tile
      const int kcl  = imin(gk, K - 1);
      const bool kok = (gk < K);
      const int edst = ((lane >> 4) << 3) + (lane & 7);  // fragment element
      const int lhal = ((lane >> 3) & 1) << 4;           // +16 if k[3]
#pragma unroll
      for (int j = 0; j < 16; ++j) {
        int gr = row0 + j;
        float v = A[(size_t)imin(gr, M - 1) * lda + kcl];
        v = (kok && gr < M) ? v : 0.0f;
        Af[(j + lhal) * 16 + edst] = (_Float16)v;
      }
    }
    // ---- stage B tile: 16 coalesced rounds, two rows of 16 floats each ----
    {
      const int gn   = col0 + (lane & 15);
      const int ncl  = imin(gn, N - 1);
      const bool nok = (gn < N);
      const int kup  = lane >> 4;              // 0/1: which of the row pair
#pragma unroll
      for (int j = 0; j < 16; ++j) {
        int kk = 2 * j + kup;                  // k within tile, 0..31
        int gk = k0 + kk;
        float v = W[(size_t)imin(gk, K - 1) * N + ncl];
        v = (nok && gk < K) ? v : 0.0f;
        Bf[((lane & 15) + (((kk >> 3) & 1) << 4)) * 16 + ((kk >> 4) << 3) + (kk & 7)] = (_Float16)v;
      }
    }
    // Prefetch next K tile of the weight matrix (global_prefetch_b8).
    if (k0 + 32 < K) {
      int pk = imin(k0 + 32 + lane, K - 1);
      __builtin_prefetch(&W[(size_t)pk * N + col0]);
    }
    __syncthreads();
    v16h af = *(const v16h*)&Af[lane * 16];   // 2x ds_load_b128
    v16h bf = *(const v16h*)&Bf[lane * 16];
    acc = __builtin_amdgcn_wmma_f32_16x16x32_f16(false, af, false, bf,
                                                 (short)0, acc, false, false);
    __syncthreads();
  }
  const int n  = lane & 15;
  const int mh = (lane >> 4) * 8;
#pragma unroll
  for (int j = 0; j < 8; ++j) {
    int gm = row0 + mh + j;
    int gn = col0 + n;
    if (gm < M && gn < N) {
      float v = acc[j];
      if (bias)  v += bias[gn];
      if (accum) v += C[(size_t)gm * N + gn];
      C[(size_t)gm * N + gn] = apply_act(v, act);
    }
  }
}

// ---------------------------------------------------------------------------
// Elementwise / utility kernels
// ---------------------------------------------------------------------------
__global__ void fill_kernel(float* p, float v, int n) {
  int i = blockIdx.x * blockDim.x + threadIdx.x;
  if (i < n) p[i] = v;
}
__global__ void act_kernel(float* p, int n, int act) {
  int i = blockIdx.x * blockDim.x + threadIdx.x;
  if (i < n) p[i] = apply_act(p[i], act);
}
__global__ void add_kernel(const float* a, const float* b, float* o, int n) {
  int i = blockIdx.x * blockDim.x + threadIdx.x;
  if (i < n) o[i] = a[i] + b[i];
}
__global__ void copy2d_kernel(const float* src, int lds, float* dst, int ldd,
                              int cols, int rows) {
  int i = blockIdx.x * blockDim.x + threadIdx.x;
  if (i >= cols * rows) return;
  int r = i / cols, c = i % cols;
  dst[(size_t)r * ldd + c] = src[(size_t)r * lds + c];
}
__global__ void cat2_kernel(const float* a, int lda, int da,
                            const float* b, int ldb, int db,
                            float* o, int rows) {
  int w = da + db;
  int i = blockIdx.x * blockDim.x + threadIdx.x;
  if (i >= rows * w) return;
  int r = i / w, c = i % w;
  o[(size_t)r * w + c] = (c < da) ? a[(size_t)r * lda + c]
                                  : b[(size_t)r * ldb + (c - da)];
}
// LayerNorm over fixed width 32, in place. One 32-thread wave per row.
__global__ void ln32_kernel(float* x, const float* g, const float* b) {
  int r = blockIdx.x, t = threadIdx.x;
  __shared__ float sm[32];
  float v = x[(size_t)r * 32 + t];
  sm[t] = v; __syncthreads();
  for (int off = 16; off > 0; off >>= 1) { if (t < off) sm[t] += sm[t + off]; __syncthreads(); }
  float mean = sm[0] * (1.0f / 32.0f); __syncthreads();
  float d = v - mean;
  sm[t] = d * d; __syncthreads();
  for (int off = 16; off > 0; off >>= 1) { if (t < off) sm[t] += sm[t + off]; __syncthreads(); }
  float var = sm[0] * (1.0f / 32.0f);
  x[(size_t)r * 32 + t] = d * rsqrtf(var + 1e-5f) * g[t] + b[t];
}
// o[n,d] = mean over the 2 tokens of x[(2n|2n+1), d]
__global__ void mean2_kernel(const float* x, float* o, int nd, int d) {
  int i = blockIdx.x * blockDim.x + threadIdx.x;
  if (i >= nd) return;
  int n = i / d, j = i % d;
  o[i] = 0.5f * (x[(size_t)(2 * n) * d + j] + x[(size_t)(2 * n + 1) * d + j]);
}
// o[(2n+s), d] = src[n, d]
__global__ void rep2_kernel(const float* src, float* o, int rd, int d) {
  int i = blockIdx.x * blockDim.x + threadIdx.x;
  if (i >= rd) return;
  int r = i / d, j = i % d;
  o[i] = src[(size_t)(r >> 1) * d + j];
}
// Seq-len-2 MHA core: d=32, 4 heads, dh=8. One thread per (node, head).
__global__ void attn22_kernel(const float* __restrict__ qq,
                              const float* __restrict__ kk,
                              const float* __restrict__ vv,
                              float* __restrict__ o, int Nn) {
  int idx = blockIdx.x * blockDim.x + threadIdx.x;
  if (idx >= Nn * 4) return;
  int n = idx >> 2, h = idx & 3;
  const float scale = 0.35355339059327373f; // 1/sqrt(8)
  for (int qp = 0; qp < 2; ++qp) {
    const float* qv = qq + (size_t)(n * 2 + qp) * 32 + h * 8;
    const float* k0 = kk + (size_t)(n * 2 + 0) * 32 + h * 8;
    const float* k1 = kk + (size_t)(n * 2 + 1) * 32 + h * 8;
    float s0 = 0.0f, s1 = 0.0f;
    for (int d = 0; d < 8; ++d) { s0 += qv[d] * k0[d]; s1 += qv[d] * k1[d]; }
    s0 *= scale; s1 *= scale;
    float mx = fmaxf(s0, s1);
    float e0 = expf(s0 - mx), e1 = expf(s1 - mx);
    float inv = 1.0f / (e0 + e1);
    float a0 = e0 * inv, a1 = e1 * inv;
    const float* v0 = vv + (size_t)(n * 2 + 0) * 32 + h * 8;
    const float* v1 = vv + (size_t)(n * 2 + 1) * 32 + h * 8;
    float* ov = o + (size_t)(n * 2 + qp) * 32 + h * 8;
    for (int d = 0; d < 8; ++d) ov[d] = a0 * v0[d] + a1 * v1[d];
  }
}
// CfC gate combine: h = ff1*(1-t) + t*ff2
__global__ void cfc_combine_kernel(const float* ff1, const float* ff2,
                                   const float* t, float* h, int n) {
  int i = blockIdx.x * blockDim.x + threadIdx.x;
  if (i >= n) return;
  float tt = t[i];
  h[i] = ff1[i] * (1.0f - tt) + tt * ff2[i];
}
// LSTM gates for set2set: z (B,4d) split i,f,g,o
__global__ void lstm_gates_kernel(const float* __restrict__ z, float* c,
                                  float* h, int B, int d) {
  int i = blockIdx.x * blockDim.x + threadIdx.x;
  if (i >= B * d) return;
  int b = i / d, j = i % d;
  const float* zr = z + (size_t)b * 4 * d;
  float gi = 1.0f / (1.0f + expf(-zr[j]));
  float gf = 1.0f / (1.0f + expf(-zr[d + j]));
  float gg = tanhf(zr[2 * d + j]);
  float go = 1.0f / (1.0f + expf(-zr[3 * d + j]));
  float cc = gf * c[i] + gi * gg;
  c[i] = cc;
  h[i] = go * tanhf(cc);
}
// Segment softmax attention for set2set. One block (d threads) per graph;
// contiguous npg nodes per graph. qstar[g] = [h[g], sum_j a_j x_j].
__global__ void s2s_attend_kernel(const float* __restrict__ x,
                                  const float* __restrict__ h,
                                  float* __restrict__ qstar,
                                  int npg, int d) {
  extern __shared__ float sm[];     // d (reduction) + npg (scores)
  float* red = sm;
  float* aa  = sm + d;
  int g = blockIdx.x, t = threadIdx.x;
  float hv = h[(size_t)g * d + t];
  int p2 = 1; while (p2 < d) p2 <<= 1;
  for (int j = 0; j < npg; ++j) {
    red[t] = x[(size_t)(g * npg + j) * d + t] * hv;
    __syncthreads();
    for (int off = p2 >> 1; off > 0; off >>= 1) {
      if (t < off && t + off < d) red[t] += red[t + off];
      __syncthreads();
    }
    if (t == 0) aa[j] = red[0];
    __syncthreads();
  }
  if (t == 0) {
    float m = aa[0];
    for (int j = 1; j < npg; ++j) m = fmaxf(m, aa[j]);
    float s = 0.0f;
    for (int j = 0; j < npg; ++j) { aa[j] = expf(aa[j] - m); s += aa[j]; }
    float inv = 1.0f / (s + 1e-16f);
    for (int j = 0; j < npg; ++j) aa[j] *= inv;
  }
  __syncthreads();
  float r = 0.0f;
  for (int j = 0; j < npg; ++j) r += aa[j] * x[(size_t)(g * npg + j) * d + t];
  qstar[(size_t)g * 2 * d + t]     = hv;
  qstar[(size_t)g * 2 * d + d + t] = r;
}
// NNConv messages + deterministic segment sum. One block per graph (edges and
// destination nodes of graph g are contiguous: edges [g*epg,(g+1)*epg),
// nodes [g*npg,(g+1)*npg) -- true by construction of the input graphs).
// msg[e,o] = sum_i x[src,i] * ( sum_h he[e,h]*w2[h,i*dout+o] + b2[i*dout+o] )
// The per-edge source row x[src,:] is staged to LDS with
// GLOBAL_LOAD_ASYNC_TO_LDS_B32 (tracked by ASYNCcnt), drained with
// s_wait_asynccnt then a workgroup barrier.
__global__ void nnconv_msg_kernel(const float* __restrict__ x, int ldx,
                                  const int* __restrict__ ei, int E,
                                  int epg, int npg,
                                  const float* __restrict__ he,
                                  const float* __restrict__ w2,
                                  const float* __restrict__ b2,
                                  float* __restrict__ sbuf,
                                  float* __restrict__ cnt,
                                  int din, int dout) {
  extern __shared__ float sm[];       // npg*dout acc | din xs | 32 he
  float* acc = sm;
  float* xs  = sm + npg * dout;
  float* hs  = xs + din;
  __shared__ float scnt[32];          // npg <= 32
  int g = blockIdx.x, t = threadIdx.x;
  for (int i = t; i < npg * dout; i += blockDim.x) acc[i] = 0.0f;
  for (int i = t; i < npg; i += blockDim.x) scnt[i] = 0.0f;
  __syncthreads();
  const int e0 = g * epg, node0 = g * npg;
  const size_t wstride = (size_t)din * dout;
  for (int k = 0; k < epg; ++k) {
    int e = e0 + k;
    int src = ei[e];
    int dst = ei[E + e];
    // Async global->LDS copy of the source node row (f32, no conversion):
    // vdst = 32-bit LDS byte address (low dword of the generic pointer),
    // vaddr = 64-bit global address pair. GV addressing mode.
    for (int i = t; i < din; i += blockDim.x) {
      unsigned lds_off = (unsigned)(uintptr_t)(&xs[i]);
      const float* gp = &x[(size_t)src * ldx + i];
      asm volatile("global_load_async_to_lds_b32 %0, %1, off"
                   :: "v"(lds_off), "v"(gp) : "memory");
    }
    for (int i = t; i < 32; i += blockDim.x) hs[i] = he[(size_t)e * 32 + i];
    asm volatile("s_wait_asynccnt 0" ::: "memory");
    __syncthreads();
    int ld = dst - node0;
    for (int o = t; o < dout; o += blockDim.x) {
      float a = 0.0f;
      for (int i = 0; i < din; ++i) {
        size_t base = (size_t)i * dout + o;
        float w = b2[base];
#pragma unroll 8
        for (int hh = 0; hh < 32; ++hh) w += hs[hh] * w2[hh * wstride + base];
        a += xs[i] * w;
      }
      acc[ld * dout + o] += a;
    }
    if (t == 0) scnt[ld] += 1.0f;
    __syncthreads();
  }
  for (int i = t; i < npg * dout; i += blockDim.x)
    sbuf[(size_t)node0 * dout + i] = acc[i];
  for (int i = t; i < npg; i += blockDim.x) cnt[node0 + i] = scnt[i];
}
// out[n,o] += sbuf[n,o] / max(cnt[n],1); optional relu
__global__ void nnconv_fin_kernel(float* out, const float* sbuf,
                                  const float* cnt, int n, int dout,
                                  int relu_out) {
  int i = blockIdx.x * blockDim.x + threadIdx.x;
  if (i >= n * dout) return;
  float v = out[i] + sbuf[i] / fmaxf(cnt[i / dout], 1.0f);
  out[i] = relu_out ? fmaxf(v, 0.0f) : v;
}
__global__ void gen_pair_edges_kernel(int* ei, int E) {
  int e = blockIdx.x * blockDim.x + threadIdx.x;
  if (e < E) { ei[e] = e; ei[E + e] = e ^ 1; }
}

// ---------------------------------------------------------------------------
// Host-side helpers
// ---------------------------------------------------------------------------
static inline void k_gemm(hipStream_t st, const float* A, int lda,
                          const float* W, const float* bias, float* C,
                          int M, int N, int K, int act, int accum) {
  dim3 grid((unsigned)((N + 15) / 16), (unsigned)((M + 15) / 16));
  gemm_wmma_kernel<<<grid, dim3(32), 0, st>>>(A, lda, W, bias, C, M, N, K, act, accum);
}
static inline void k_fill(hipStream_t st, float* p, float v, int n) {
  fill_kernel<<<(n + 255) / 256, 256, 0, st>>>(p, v, n);
}
static inline void k_act(hipStream_t st, float* p, int n, int act) {
  act_kernel<<<(n + 255) / 256, 256, 0, st>>>(p, n, act);
}
static inline void k_add(hipStream_t st, const float* a, const float* b, float* o, int n) {
  add_kernel<<<(n + 255) / 256, 256, 0, st>>>(a, b, o, n);
}
static inline void k_copy2d(hipStream_t st, const float* s, int lds, float* d,
                            int ldd, int cols, int rows) {
  copy2d_kernel<<<(cols * rows + 255) / 256, 256, 0, st>>>(s, lds, d, ldd, cols, rows);
}
static inline void k_cat2(hipStream_t st, const float* a, int lda, int da,
                          const float* b, int ldb, int db, float* o, int rows) {
  cat2_kernel<<<((da + db) * rows + 255) / 256, 256, 0, st>>>(a, lda, da, b, ldb, db, o, rows);
}
static inline void k_ln32(hipStream_t st, float* x, const float* g, const float* b, int rows) {
  ln32_kernel<<<rows, 32, 0, st>>>(x, g, b);
}

// Parameter flat-index map (depth-first insertion order of make_params()).
enum ParamIdx {
  IN_X1S = 0, IN_EI1, IN_EA1, IN_B1, IN_X2S, IN_EI2, IN_EA2, IN_B2, IN_GEA, IN_NG,
  A11_W1 = 10, A11_B1, A11_W2, A11_B2, A11_ROOT, A11_BIAS,
  A21_W1 = 16, A21_B1, A21_W2, A21_B2, A21_ROOT, A21_BIAS,
  SC1_W1 = 22, SC1_B1, SC1_W2, SC1_B2, SC1_ROOT, SC1_BIAS,
  SC2_W1 = 28, SC2_B1, SC2_W2, SC2_B2, SC2_ROOT, SC2_BIAS,
  GC_W1  = 34, GC_B1,  GC_W2,  GC_B2,  GC_ROOT,  GC_BIAS,
  NCP_WF1 = 40, NCP_BF1, NCP_WF2, NCP_BF2, NCP_WTA, NCP_BTA, NCP_WTB, NCP_BTB,
  XM2_WF1 = 48, XM2_BF1, XM2_WF2, XM2_BF2, XM2_WTA, XM2_BTA, XM2_WTB, XM2_BTB,
  XM2_WB0 = 56, XM2_B0,
  X22_W = 58, X22_B,
  LAST_W = 60, LAST_B,
  ENC0 = 62,   // +0 wq,+1 bq,+2 wk,+3 bk,+4 wv,+5 bv,+6 wo,+7 bo,
               // +8 ln1g,+9 ln1b,+10 wf1,+11 bf1,+12 wf2,+13 bf2,+14 ln2g,+15 ln2b
  ENC1 = 78,
  DEC0 = 94,   // +0..7 sa, +8..15 ca, +16 ln1g,+17 ln1b,+18 ln2g,+19 ln2b,
               // +20 wf1,+21 bf1,+22 wf2,+23 bf2,+24 ln3g,+25 ln3b
  DEC1 = 120,
  S2S_WIH = 146, S2S_WHH, S2S_B,
  S2S2_WIH = 149, S2S2_WHH, S2S2_B,
  FF_W = 152, FF_B,
  FC0_W = 154, FC0_B, FC1_W, FC1_B, FC2_W, FC2_B, FC3_W, FC3_B
};
#define PF(i) ((const float*)din[(i)])

struct Ws {
  float *he, *sbuf, *cnt;
  float *x1, *h12, *z18, *ff1, *ff2, *tb, *y, *y2;
  float *mem, *qq, *kk, *vv, *ao, *mo, *ff64, *meanb, *dec;
  float *xm, *h2, *z128, *bb, *fd1, *fd2, *td, *o2, *zcat, *zlast, *sA, *sB;
  float *h128, *c128, *z512, *s1, *s2;
  float *x2b2, *x2out2, *tout2;
  float *ex, *heg, *sgc, *cntg, *cbuf, *comb;
  float *qs2, *h384, *c384, *z1536, *ffB, *hc, *f1, *f2, *f3;
  int *eig;
};

static void nnconv_run(hipStream_t st, const float* x, int ldx, int din, int dout,
                       int Nn, const int* ei, int E, int epg, int npg,
                       const float* ea, int ed,
                       const float* w1, const float* b1, const float* w2,
                       const float* b2, const float* root, const float* bias,
                       float* out, int relu_out, Ws& S) {
  // he = relu(ea @ w1 + b1)   (E x 32)
  k_gemm(st, ea, ed, w1, b1, S.he, E, 32, ed, ACT_RELU, 0);
  int blocks = E / epg;
  int blk = dout < 32 ? 32 : dout;
  size_t shmem = (size_t)(npg * dout + din + 32) * sizeof(float);
  nnconv_msg_kernel<<<blocks, blk, shmem, st>>>(x, ldx, ei, E, epg, npg, S.he,
                                                w2, b2, S.sbuf, S.cnt, din, dout);
  // out = x @ root + bias, then += s/max(cnt,1) (+relu)
  k_gemm(st, x, ldx, root, bias, out, Nn, dout, din, ACT_NONE, 0);
  nnconv_fin_kernel<<<(Nn * dout + 255) / 256, 256, 0, st>>>(out, S.sbuf, S.cnt,
                                                             Nn, dout, relu_out);
}

static void mha_run(hipStream_t st, void* const* din, int pb, const float* qin,
                    const float* kvin, int R, float* out, Ws& S) {
  k_gemm(st, qin, 32, PF(pb + 0), PF(pb + 1), S.qq, R, 32, 32, ACT_NONE, 0);
  k_gemm(st, kvin, 32, PF(pb + 2), PF(pb + 3), S.kk, R, 32, 32, ACT_NONE, 0);
  k_gemm(st, kvin, 32, PF(pb + 4), PF(pb + 5), S.vv, R, 32, 32, ACT_NONE, 0);
  int Nn = R / 2;
  attn22_kernel<<<(Nn * 4 + 255) / 256, 256, 0, st>>>(S.qq, S.kk, S.vv, S.ao, Nn);
  k_gemm(st, S.ao, 32, PF(pb + 6), PF(pb + 7), out, R, 32, 32, ACT_NONE, 0);
}

static void enc_layer_run(hipStream_t st, void* const* din, int pb, float* mem,
                          int R, Ws& S) {
  mha_run(st, din, pb + 0, mem, mem, R, S.mo, S);
  k_add(st, mem, S.mo, mem, R * 32);
  k_ln32(st, mem, PF(pb + 8), PF(pb + 9), R);
  k_gemm(st, mem, 32, PF(pb + 10), PF(pb + 11), S.ff64, R, 64, 32, ACT_RELU, 0);
  k_gemm(st, S.ff64, 64, PF(pb + 12), PF(pb + 13), S.mo, R, 32, 64, ACT_NONE, 0);
  k_add(st, mem, S.mo, mem, R * 32);
  k_ln32(st, mem, PF(pb + 14), PF(pb + 15), R);
}

static void dec_layer_run(hipStream_t st, void* const* din, int pb, float* t,
                          const float* mem, int R, Ws& S) {
  mha_run(st, din, pb + 0, t, t, R, S.mo, S);
  k_add(st, t, S.mo, t, R * 32);
  k_ln32(st, t, PF(pb + 16), PF(pb + 17), R);
  mha_run(st, din, pb + 8, t, mem, R, S.mo, S);
  k_add(st, t, S.mo, t, R * 32);
  k_ln32(st, t, PF(pb + 18), PF(pb + 19), R);
  k_gemm(st, t, 32, PF(pb + 20), PF(pb + 21), S.ff64, R, 64, 32, ACT_RELU, 0);
  k_gemm(st, S.ff64, 64, PF(pb + 22), PF(pb + 23), S.mo, R, 32, 64, ACT_NONE, 0);
  k_add(st, t, S.mo, t, R * 32);
  k_ln32(st, t, PF(pb + 24), PF(pb + 25), R);
}

static void set2set_run(hipStream_t st, const float* x, int npg, int B, int d,
                        const float* wih, const float* whh, const float* bvec,
                        float* qstar, float* h, float* c, float* z) {
  k_fill(st, qstar, 0.0f, B * 2 * d);
  k_fill(st, h, 0.0f, B * d);
  k_fill(st, c, 0.0f, B * d);
  for (int step = 0; step < 2; ++step) {
    k_gemm(st, qstar, 2 * d, wih, bvec, z, B, 4 * d, 2 * d, ACT_NONE, 0);
    k_gemm(st, h, d, whh, nullptr, z, B, 4 * d, d, ACT_NONE, 1);
    lstm_gates_kernel<<<(B * d + 255) / 256, 256, 0, st>>>(z, c, h, B, d);
    s2s_attend_kernel<<<B, d, (size_t)(d + npg) * sizeof(float), st>>>(x, h, qstar, npg, d);
  }
}

static void run_subgraph(hipStream_t st, void* const* din, const float* x,
                         const int* ei, const float* ea, int N, int E, int B,
                         int npg, int epg, Ws& S, float* x2dest,
                         float* x2outdest, float* toutdest, float* s_out) {
  // x1 = relu(nnconv(x[:, :41], a11))
  nnconv_run(st, x, 47, 41, 32, N, ei, E, epg, npg, ea, 10,
             PF(A11_W1), PF(A11_B1), PF(A11_W2), PF(A11_B2),
             PF(A11_ROOT), PF(A11_BIAS), S.x1, 1, S);
  // x2 = x[:, 41:]
  k_copy2d(st, x + 41, 47, x2dest, 6, 6, N);
  // CfC-NCP loop (5 iters), h = [x2, x2]
  k_cat2(st, x2dest, 6, 6, x2dest, 6, 6, S.h12, N);
  for (int it = 0; it < 5; ++it) {
    k_cat2(st, x2dest, 6, 6, S.h12, 12, 12, S.z18, N);
    k_gemm(st, S.z18, 18, PF(NCP_WF1), PF(NCP_BF1), S.ff1, N, 12, 18, ACT_TANH, 0);
    k_gemm(st, S.z18, 18, PF(NCP_WF2), PF(NCP_BF2), S.ff2, N, 12, 18, ACT_TANH, 0);
    k_gemm(st, S.z18, 18, PF(NCP_WTA), PF(NCP_BTA), S.tb, N, 12, 18, ACT_NONE, 0);
    k_gemm(st, S.z18, 18, PF(NCP_WTB), PF(NCP_BTB), S.tb, N, 12, 18, ACT_NONE, 1);
    k_act(st, S.tb, N * 12, ACT_SIG);
    cfc_combine_kernel<<<(N * 12 + 255) / 256, 256, 0, st>>>(S.ff1, S.ff2, S.tb, S.h12, N * 12);
    k_copy2d(st, S.h12, 12, x2outdest + 6 * it, 30, 6, N);  // o = h[:, :6]
  }
  // y = relu(x2out30 @ x22); y = relu(nnconv(y, a21))
  k_gemm(st, x2outdest, 30, PF(X22_W), PF(X22_B), S.y, N, 32, 30, ACT_RELU, 0);
  nnconv_run(st, S.y, 32, 32, 32, N, ei, E, epg, npg, ea, 10,
             PF(A21_W1), PF(A21_B1), PF(A21_W2), PF(A21_B2),
             PF(A21_ROOT), PF(A21_BIAS), S.y2, 1, S);
  // mem = stack([x1, y], axis=1) as (2N, 32) token rows
  k_copy2d(st, S.x1, 32, S.mem, 64, 32, N);
  k_copy2d(st, S.y2, 32, S.mem + 32, 64, 32, N);
  const int R = 2 * N;
  enc_layer_run(st, din, ENC0, S.mem, R, S);
  enc_layer_run(st, din, ENC1, S.mem, R, S);
  // tgt = repeat(mean(mem,1), 2)
  mean2_kernel<<<(N * 32 + 255) / 256, 256, 0, st>>>(S.mem, S.meanb, N * 32, 32);
  rep2_kernel<<<(R * 32 + 255) / 256, 256, 0, st>>>(S.meanb, S.dec, R * 32, 32);
  dec_layer_run(st, din, DEC0, S.dec, S.mem, R, S);
  dec_layer_run(st, din, DEC1, S.dec, S.mem, R, S);
  // tout = relu(mean(dec,1))
  mean2_kernel<<<(N * 32 + 255) / 256, 256, 0, st>>>(S.dec, toutdest, N * 32, 32);
  k_act(st, toutdest, N * 32, ACT_RELU);
  // cfc_dense(xm=[x1,y], h2=[tout,tout])
  k_cat2(st, S.x1, 32, 32, S.y2, 32, 32, S.xm, N);
  k_cat2(st, toutdest, 32, 32, toutdest, 32, 32, S.h2, N);
  k_cat2(st, S.xm, 64, 64, S.h2, 64, 64, S.z128, N);
  k_gemm(st, S.z128, 128, PF(XM2_WB0), PF(XM2_B0), S.bb, N, 128, 128, ACT_LECUN, 0);
  k_gemm(st, S.bb, 128, PF(XM2_WF1), PF(XM2_BF1), S.fd1, N, 64, 128, ACT_TANH, 0);
  k_gemm(st, S.bb, 128, PF(XM2_WF2), PF(XM2_BF2), S.fd2, N, 64, 128, ACT_TANH, 0);
  k_gemm(st, S.bb, 128, PF(XM2_WTA), PF(XM2_BTA), S.td, N, 64, 128, ACT_NONE, 0);
  k_gemm(st, S.bb, 128, PF(XM2_WTB), PF(XM2_BTB), S.td, N, 64, 128, ACT_NONE, 1);
  k_act(st, S.td, N * 64, ACT_SIG);
  cfc_combine_kernel<<<(N * 64 + 255) / 256, 256, 0, st>>>(S.fd1, S.fd2, S.td, S.o2, N * 64);
  // z = relu(o2); z = [z, xm]; z = relu(z @ last)
  k_act(st, S.o2, N * 64, ACT_RELU);
  k_cat2(st, S.o2, 64, 64, S.xm, 64, 64, S.zcat, N);
  k_gemm(st, S.zcat, 128, PF(LAST_W), PF(LAST_B), S.zlast, N, 128, 128, ACT_RELU, 0);
  // s = relu(sc1); s = relu(sc2); s = sc2 (no relu)
  nnconv_run(st, S.zlast, 128, 128, 128, N, ei, E, epg, npg, ea, 10,
             PF(SC1_W1), PF(SC1_B1), PF(SC1_W2), PF(SC1_B2),
             PF(SC1_ROOT), PF(SC1_BIAS), S.sA, 1, S);
  nnconv_run(st, S.sA, 128, 128, 128, N, ei, E, epg, npg, ea, 10,
             PF(SC2_W1), PF(SC2_B1), PF(SC2_W2), PF(SC2_B2),
             PF(SC2_ROOT), PF(SC2_BIAS), S.sB, 1, S);
  nnconv_run(st, S.sB, 128, 128, 128, N, ei, E, epg, npg, ea, 10,
             PF(SC2_W1), PF(SC2_B1), PF(SC2_W2), PF(SC2_B2),
             PF(SC2_ROOT), PF(SC2_BIAS), S.sA, 0, S);
  // s = set2set(s) -> (B, 256)
  set2set_run(st, S.sA, npg, B, 128, PF(S2S_WIH), PF(S2S_WHH), PF(S2S_B),
              s_out, S.h128, S.c128, S.z512);
}

// ---------------------------------------------------------------------------
extern "C" void kernel_launch(void* const* d_in, const int* in_sizes, int n_in,
                              void* d_out, int out_size, void* d_ws, size_t ws_size,
                              hipStream_t stream) {
  void* const* din = d_in;
  const int N = in_sizes[IN_X1S] / 47;   // 2048 nodes per subgraph
  const int E = in_sizes[IN_EI1] / 2;    // 4096 edges per subgraph
  const int B = in_sizes[IN_GEA] / 8;    // gea is (2B, 4) -> B = 256
  const int npg = N / B, epg = E / B;
  float* fout = (float*)d_out;

  // Bump-allocate scratch (floats) out of d_ws (~24 MB total).
  float* wsf = (float*)d_ws;
  size_t off = 0;
  auto alloc = [&](size_t n) { float* p = wsf + off; off += (n + 63) & ~(size_t)63; return p; };
  Ws S;
  S.he   = alloc((size_t)E * 32);   S.sbuf = alloc((size_t)N * 128); S.cnt = alloc(N);
  S.x1   = alloc((size_t)N * 32);   S.h12  = alloc((size_t)N * 12);
  S.z18  = alloc((size_t)N * 18);   S.ff1  = alloc((size_t)N * 12);
  S.ff2  = alloc((size_t)N * 12);   S.tb   = alloc((size_t)N * 12);
  S.y    = alloc((size_t)N * 32);   S.y2   = alloc((size_t)N * 32);
  S.mem  = alloc((size_t)N * 64);   S.qq   = alloc((size_t)N * 64);
  S.kk   = alloc((size_t)N * 64);   S.vv   = alloc((size_t)N * 64);
  S.ao   = alloc((size_t)N * 64);   S.mo   = alloc((size_t)N * 64);
  S.ff64 = alloc((size_t)N * 128);  S.meanb= alloc((size_t)N * 32);
  S.dec  = alloc((size_t)N * 64);
  S.xm   = alloc((size_t)N * 64);   S.h2   = alloc((size_t)N * 64);
  S.z128 = alloc((size_t)N * 128);  S.bb   = alloc((size_t)N * 128);
  S.fd1  = alloc((size_t)N * 64);   S.fd2  = alloc((size_t)N * 64);
  S.td   = alloc((size_t)N * 64);   S.o2   = alloc((size_t)N * 64);
  S.zcat = alloc((size_t)N * 128);  S.zlast= alloc((size_t)N * 128);
  S.sA   = alloc((size_t)N * 128);  S.sB   = alloc((size_t)N * 128);
  S.h128 = alloc((size_t)B * 128);  S.c128 = alloc((size_t)B * 128);
  S.z512 = alloc((size_t)B * 512);
  S.s1   = alloc((size_t)B * 256);  S.s2   = alloc((size_t)B * 256);
  S.x2b2 = alloc((size_t)N * 6);    S.x2out2 = alloc((size_t)N * 30);
  S.tout2= alloc((size_t)N * 32);
  S.ex   = alloc((size_t)2 * B * 256);
  S.heg  = alloc((size_t)2 * B * 32);
  S.sgc  = alloc((size_t)2 * B * 128); S.cntg = alloc((size_t)2 * B);
  S.cbuf = alloc((size_t)2 * B * 128); S.comb = alloc((size_t)2 * B * 384);
  S.qs2  = alloc((size_t)B * 768);  S.h384 = alloc((size_t)B * 384);
  S.c384 = alloc((size_t)B * 384);  S.z1536= alloc((size_t)B * 1536);
  S.ffB  = alloc((size_t)B * 64);   S.hc   = alloc((size_t)B * 320);
  S.f1   = alloc((size_t)B * 512);  S.f2   = alloc((size_t)B * 128);
  S.f3   = alloc((size_t)B * 64);
  S.eig  = (int*)alloc((size_t)4 * B + 64);
  (void)n_in; (void)out_size; (void)ws_size;

  // Output layout: [B] h | [N*6] x11 | [N*30] x2out | [N*32] tout
  float* out_h     = fout;
  float* out_x11   = fout + B;
  float* out_x2out = out_x11 + (size_t)N * 6;
  float* out_tout  = out_x2out + (size_t)N * 30;

  // Subgraph 1 (its x2/x2out/tout are part of the output tuple).
  run_subgraph(stream, din, PF(IN_X1S), (const int*)din[IN_EI1], PF(IN_EA1),
               N, E, B, npg, epg, S, out_x11, out_x2out, out_tout, S.s1);
  // Subgraph 2.
  run_subgraph(stream, din, PF(IN_X2S), (const int*)din[IN_EI2], PF(IN_EA2),
               N, E, B, npg, epg, S, S.x2b2, S.x2out2, S.tout2, S.s2);

  // ex = relu(stack([s1, s2], 1)).reshape(2B, 256)
  k_act(stream, S.s1, B * 256, ACT_RELU);
  k_act(stream, S.s2, B * 256, ACT_RELU);
  k_copy2d(stream, S.s1, 256, S.ex, 512, 256, B);        // rows 0,2,4,...
  k_copy2d(stream, S.s2, 256, S.ex + 256, 512, 256, B);  // rows 1,3,5,...
  // ei_g = [idx, idx^1]; gc conv on pair-graphs (epg=2, npg=2 per pair)
  gen_pair_edges_kernel<<<(2 * B + 255) / 256, 256, 0, stream>>>(S.eig, 2 * B);
  {
    Ws G = S; G.he = S.heg; G.sbuf = S.sgc; G.cnt = S.cntg;
    nnconv_run(stream, S.ex, 256, 256, 128, 2 * B, S.eig, 2 * B, 2, 2,
               PF(IN_GEA), 4, PF(GC_W1), PF(GC_B1), PF(GC_W2), PF(GC_B2),
               PF(GC_ROOT), PF(GC_BIAS), S.cbuf, 1, G);
  }
  // comb = [c, ex]; q = set2set(comb, 2 nodes per segment, d=384)
  k_cat2(stream, S.cbuf, 128, 128, S.ex, 256, 256, S.comb, 2 * B);
  set2set_run(stream, S.comb, 2, B, 384, PF(S2S2_WIH), PF(S2S2_WHH), PF(S2S2_B),
              S.qs2, S.h384, S.c384, S.z1536);
  // ff = relu(q @ FF); head = [ex[0::2], ff]; fc chain -> (B, 1)
  k_gemm(stream, S.qs2, 768, PF(FF_W), PF(FF_B), S.ffB, B, 64, 768, ACT_RELU, 0);
  k_cat2(stream, S.ex, 512, 256, S.ffB, 64, 64, S.hc, B);
  k_gemm(stream, S.hc, 320, PF(FC0_W), PF(FC0_B), S.f1, B, 512, 320, ACT_RELU, 0);
  k_gemm(stream, S.f1, 512, PF(FC1_W), PF(FC1_B), S.f2, B, 128, 512, ACT_RELU, 0);
  k_gemm(stream, S.f2, 128, PF(FC2_W), PF(FC2_B), S.f3, B, 64, 128, ACT_RELU, 0);
  k_gemm(stream, S.f3, 64, PF(FC3_W), PF(FC3_B), out_h, B, 1, 64, ACT_NONE, 0);
}
